// CAGN_52304111730951
// MI455X (gfx1250) — matrix operand order repaired and checked by reading, not compiled, not run on hardware
//
#include <hip/hip_runtime.h>
#include <hip/hip_bf16.h>
#include <math.h>

#define F_IN   128
#define HIDC   32
#define HEADS  4
#define OUTC   10
#define NSLOPE 0.2f

typedef __bf16 bf16x16 __attribute__((ext_vector_type(16)));
typedef float  f32x8   __attribute__((ext_vector_type(8)));

union Frag16 { bf16x16 v; unsigned u[8]; };

__device__ __forceinline__ unsigned short f2bf(float f) {
    unsigned u = __float_as_uint(f);
    unsigned r = u + 0x7FFFu + ((u >> 16) & 1u);   // round-to-nearest-even
    return (unsigned short)(r >> 16);
}
// order-preserving float->uint map for atomicMax-based segment max
__device__ __forceinline__ unsigned f2o(float f) {
    unsigned u = __float_as_uint(f);
    return (u & 0x80000000u) ? ~u : (u | 0x80000000u);
}
__device__ __forceinline__ float o2f(unsigned u) {
    return (u & 0x80000000u) ? __uint_as_float(u & 0x7FFFFFFFu) : __uint_as_float(~u);
}

// A fragment: 16x32 bf16, row-major source [rows][kdim].
// lane L: m = L&15, half = L>>4; element e: K = k0 + 16*(e>>3) + 8*half + (e&7)
__device__ __forceinline__ bf16x16 load_fragA(const unsigned short* __restrict__ base,
                                              int row, int kdim, int k0, int lane) {
    Frag16 f;
    const unsigned short* p = base + (size_t)row * kdim + k0 + 8 * (lane >> 4);
#pragma unroll
    for (int v = 0; v < 8; ++v) {
        int k = 16 * (v >> 2) + 2 * (v & 3);
        f.u[v] = *(const unsigned*)(p + k);
    }
    return f.v;
}
// B fragment: 32x16 bf16, source pre-transposed [ncols][kdim].
// lane L: n = L&15, half = L>>4; element e: K = k0 + 16*(e>>3) + 8*half + (e&7)
__device__ __forceinline__ bf16x16 load_fragB(const unsigned short* __restrict__ base,
                                              int col, int kdim, int k0, int lane) {
    Frag16 f;
    const unsigned short* p = base + (size_t)col * kdim + k0 + 8 * (lane >> 4);
#pragma unroll
    for (int v = 0; v < 8; ++v) f.u[v] = *(const unsigned*)(p + 16 * (v >> 2) + 2 * (v & 3));
    return f.v;
}

// C[M,NVALID] = A[M,KDIM](bf16) @ Bt[NCOLS,KDIM]^T(bf16), f32 acc via WMMA.
// LDC/NVALID compile-time -> epilogue is 8 immediate-offset stores, no muls.
// FULL=true  -> Mrows % 16 == 0, no row guards (EXEC stays all-ones throughout).
template <int KDIM, int NCOLS, int LDC, int NVALID, bool FULL>
__global__ void k_wmma_gemm(const unsigned short* __restrict__ A,
                            const unsigned short* __restrict__ Bt,
                            float* __restrict__ C, int Mrows) {
    const int lane   = threadIdx.x & 31;
    const int wid    = blockIdx.x * (blockDim.x >> 5) + (threadIdx.x >> 5);
    const int tilesN = NCOLS / 16;
    const int Mtiles = (Mrows + 15) >> 4;
    const int mtile  = wid / tilesN;
    const int ntile  = wid % tilesN;
    if (mtile >= Mtiles) return;               // wave-uniform exit
    int row = mtile * 16 + (lane & 15);
    if (!FULL && row >= Mrows) row = Mrows - 1;  // address clamp only
    const int col = ntile * 16 + (lane & 15);
    f32x8 acc = {};
#pragma unroll
    for (int k0 = 0; k0 < KDIM; k0 += 32) {
        bf16x16 a = load_fragA(A, row, KDIM, k0, lane);
        bf16x16 b = load_fragB(Bt, col, KDIM, k0, lane);
        acc = __builtin_amdgcn_wmma_f32_16x16x32_bf16(false, a, false, b,
                                                      (short)0, acc, false, false);
    }
    const int n  = ntile * 16 + (lane & 15);
    const int m0 = mtile * 16 + 8 * (lane >> 4);
    if (n < NVALID) {
        float* cp = C + (size_t)m0 * LDC + n;
#pragma unroll
        for (int r = 0; r < 8; ++r) {
            if (FULL || (m0 + r) < Mrows) cp[(size_t)r * LDC] = acc[r];
        }
    }
}

template <int KDIM, int NCOLS, int LDC, int NVALID>
static void launch_gemm(const unsigned short* A, const unsigned short* Bt, float* C,
                        int Mrows, hipStream_t stream) {
    const int Mtiles = (Mrows + 15) / 16;
    const long long waves = (long long)Mtiles * (NCOLS / 16);
    const int grid = (int)((waves * 32 + 255) / 256);
    if (Mrows % 16 == 0)
        k_wmma_gemm<KDIM, NCOLS, LDC, NVALID, true><<<grid, 256, 0, stream>>>(A, Bt, C, Mrows);
    else
        k_wmma_gemm<KDIM, NCOLS, LDC, NVALID, false><<<grid, 256, 0, stream>>>(A, Bt, C, Mrows);
}

__global__ void k_cvt_bf16(const float* __restrict__ s, unsigned short* __restrict__ d, int n) {
    int i = blockIdx.x * blockDim.x + threadIdx.x, st = gridDim.x * blockDim.x;
    for (; i < n; i += st) d[i] = f2bf(s[i]);
}

// W[K,Nv] (row-major) -> Wt[Np,K] bf16, zero-padded columns n>=Nv
__global__ void k_transpose_cvt(const float* __restrict__ W, unsigned short* __restrict__ Wt,
                                int K, int Nv, int Np) {
    int i = blockIdx.x * blockDim.x + threadIdx.x;
    if (i >= K * Np) return;
    int n = i / K, k = i % K;
    Wt[i] = (n < Nv) ? f2bf(W[k * Nv + n]) : (unsigned short)0;
}

// per-node attention logits: os/od[n*H+h] = dot(h[n,h,:], a_{s,d}[h,:])
__global__ void k_alpha(const float* __restrict__ h, const float* __restrict__ avs,
                        const float* __restrict__ avd, float* __restrict__ os,
                        float* __restrict__ od, int Nn, int H, int C) {
    int i = blockIdx.x * blockDim.x + threadIdx.x;
    if (i >= Nn * H) return;
    int n = i / H, hh = i % H;
    const float* hp = h + (size_t)n * H * C + hh * C;
    float s = 0.f, d = 0.f;
    for (int c = 0; c < C; ++c) { float v = hp[c]; s += v * avs[hh * C + c]; d += v * avd[hh * C + c]; }
    os[i] = s; od[i] = d;
}

__global__ void k_init_u32(unsigned* __restrict__ p, unsigned v, int n) {
    int i = blockIdx.x * blockDim.x + threadIdx.x;
    if (i < n) p[i] = v;
}

__device__ __forceinline__ void edge_sd(int i, const int* src, const int* dst, int E,
                                        int& s, int& d) {
    if (i < E) { s = src[i]; d = dst[i]; } else { s = i - E; d = i - E; }
}

__global__ void k_edge_max(const int* __restrict__ src, const int* __restrict__ dst,
                           int E, int Nn, int H, const float* __restrict__ as,
                           const float* __restrict__ ad, unsigned* __restrict__ emax) {
    int i = blockIdx.x * blockDim.x + threadIdx.x, st = gridDim.x * blockDim.x;
    int total = (E + Nn) * H;
    for (; i < total; i += st) {
        int e = i / H, hh = i % H, s, d;
        edge_sd(e, src, dst, E, s, d);
        float v = as[s * H + hh] + ad[d * H + hh];
        v = v > 0.f ? v : NSLOPE * v;
        atomicMax(&emax[d * H + hh], f2o(v));
    }
}

__global__ void k_edge_sum(const int* __restrict__ src, const int* __restrict__ dst,
                           int E, int Nn, int H, const float* __restrict__ as,
                           const float* __restrict__ ad, const unsigned* __restrict__ emax,
                           float* __restrict__ denom) {
    int i = blockIdx.x * blockDim.x + threadIdx.x, st = gridDim.x * blockDim.x;
    int total = (E + Nn) * H;
    for (; i < total; i += st) {
        int e = i / H, hh = i % H, s, d;
        edge_sd(e, src, dst, E, s, d);
        float v = as[s * H + hh] + ad[d * H + hh];
        v = v > 0.f ? v : NSLOPE * v;
        atomicAdd(&denom[d * H + hh], __expf(v - o2f(emax[d * H + hh])));
    }
}

// one wave32 per edge; lanes stride the H*C channels; scatter-add into out[dst].
// C compile-time: idx/C becomes a shift (32) or constant 0 (10).
template <int C>
__global__ void k_edge_agg(const int* __restrict__ src, const int* __restrict__ dst,
                           int E, int Nn, int H, const float* __restrict__ as,
                           const float* __restrict__ ad, const unsigned* __restrict__ emax,
                           const float* __restrict__ denom, const float* __restrict__ h,
                           float* __restrict__ out) {
    const int lane = threadIdx.x & 31;
    int wid = blockIdx.x * (blockDim.x >> 5) + (threadIdx.x >> 5);
    const int nw = gridDim.x * (blockDim.x >> 5);
    const int ET = E + Nn, HC = H * C;
    for (; wid < ET; wid += nw) {
        int s, d;
        edge_sd(wid, src, dst, E, s, d);
        for (int idx = lane; idx < HC; idx += 32) {
            const int hh = idx / C;
            float v = as[s * H + hh] + ad[d * H + hh];
            v = v > 0.f ? v : NSLOPE * v;
            float a = __expf(v - o2f(emax[d * H + hh])) / (denom[d * H + hh] + 1e-16f);
            atomicAdd(&out[(size_t)d * HC + idx], h[(size_t)s * HC + idx] * a);
        }
    }
}

// out = elu(out + bias), also emit bf16 copy for the next layer's WMMA GEMM
__global__ void k_fin_elu_bf(float* __restrict__ out, const float* __restrict__ b,
                             unsigned short* __restrict__ obf, int Nn, int CH) {
    int i = blockIdx.x * blockDim.x + threadIdx.x, st = gridDim.x * blockDim.x;
    int total = Nn * CH;
    for (; i < total; i += st) {
        float v = out[i] + b[i % CH];
        v = v > 0.f ? v : (__expf(v) - 1.f);
        out[i] = v;
        obf[i] = f2bf(v);
    }
}

__global__ void k_fin_bias(float* __restrict__ out, const float* __restrict__ b, int Nn, int CH) {
    int i = blockIdx.x * blockDim.x + threadIdx.x;
    if (i < Nn * CH) out[i] += b[i % CH];
}

static inline int nblk(long long n, int b) { return (int)((n + b - 1) / b); }

extern "C" void kernel_launch(void* const* d_in, const int* in_sizes, int n_in,
                              void* d_out, int out_size, void* d_ws, size_t ws_size,
                              hipStream_t stream) {
    const float* x   = (const float*)d_in[0];
    const int*   ei  = (const int*)d_in[1];
    const float* W1  = (const float*)d_in[2];
    const float* as1 = (const float*)d_in[3];
    const float* ad1 = (const float*)d_in[4];
    const float* b1  = (const float*)d_in[5];
    const float* W2  = (const float*)d_in[6];
    const float* as2 = (const float*)d_in[7];
    const float* ad2 = (const float*)d_in[8];
    const float* b2  = (const float*)d_in[9];
    const float* W3  = (const float*)d_in[10];
    const float* as3 = (const float*)d_in[11];
    const float* ad3 = (const float*)d_in[12];
    const float* b3  = (const float*)d_in[13];

    const int Nn = in_sizes[0] / F_IN;
    const int E  = in_sizes[1] / 2;
    const int ET = E + Nn;
    const int* srcv = ei;
    const int* dstv = ei + E;

    char* p = (char*)d_ws;
    auto carve = [&](size_t bytes) { char* r = p; p += (bytes + 255) & ~(size_t)255; return r; };

    unsigned short* xb   = (unsigned short*)carve((size_t)Nn * F_IN * 2);
    unsigned short* W1t  = (unsigned short*)carve((size_t)F_IN * F_IN * 2);
    float*          h1   = (float*)carve((size_t)Nn * F_IN * 4);
    float*          aS1  = (float*)carve((size_t)Nn * HEADS * 4);
    float*          aD1  = (float*)carve((size_t)Nn * HEADS * 4);
    unsigned*       em1  = (unsigned*)carve((size_t)Nn * HEADS * 4);
    float*          dn1  = (float*)carve((size_t)Nn * HEADS * 4);
    float*          o1   = (float*)carve((size_t)Nn * F_IN * 4);
    unsigned short* h1b  = (unsigned short*)carve((size_t)Nn * F_IN * 2);
    unsigned short* W2t  = (unsigned short*)carve((size_t)HIDC * F_IN * 2);
    float*          z2   = (float*)carve((size_t)Nn * HIDC * 4);
    float*          aS2  = (float*)carve((size_t)Nn * 4);
    float*          aD2  = (float*)carve((size_t)Nn * 4);
    unsigned*       em2  = (unsigned*)carve((size_t)Nn * 4);
    float*          dn2  = (float*)carve((size_t)Nn * 4);
    float*          o2   = (float*)carve((size_t)Nn * HIDC * 4);
    unsigned short* z2b  = (unsigned short*)carve((size_t)Nn * HIDC * 2);
    unsigned short* W3t  = (unsigned short*)carve((size_t)16 * HIDC * 2);
    float*          h3   = (float*)carve((size_t)Nn * OUTC * 4);
    float*          aS3  = (float*)carve((size_t)Nn * 4);
    float*          aD3  = (float*)carve((size_t)Nn * 4);
    unsigned*       em3  = (unsigned*)carve((size_t)Nn * 4);
    float*          dn3  = (float*)carve((size_t)Nn * 4);
    float*          outf = (float*)d_out;

    const unsigned NEGINF_ORD = 0x007FFFFFu;   // f2o(-inf)

    // ---------------- Layer 1: H=4, C=32, concat ----------------
    k_cvt_bf16<<<nblk((long long)Nn * F_IN, 256), 256, 0, stream>>>(x, xb, Nn * F_IN);
    k_transpose_cvt<<<nblk(F_IN * F_IN, 256), 256, 0, stream>>>(W1, W1t, F_IN, F_IN, F_IN);
    launch_gemm<F_IN, 128, 128, 128>(xb, W1t, h1, Nn, stream);
    k_alpha<<<nblk((long long)Nn * HEADS, 256), 256, 0, stream>>>(h1, as1, ad1, aS1, aD1, Nn, HEADS, HIDC);
    k_init_u32<<<nblk((long long)Nn * HEADS, 256), 256, 0, stream>>>(em1, NEGINF_ORD, Nn * HEADS);
    k_edge_max<<<nblk((long long)ET * HEADS, 256), 256, 0, stream>>>(srcv, dstv, E, Nn, HEADS, aS1, aD1, em1);
    hipMemsetAsync(dn1, 0, (size_t)Nn * HEADS * 4, stream);
    k_edge_sum<<<nblk((long long)ET * HEADS, 256), 256, 0, stream>>>(srcv, dstv, E, Nn, HEADS, aS1, aD1, em1, dn1);
    hipMemsetAsync(o1, 0, (size_t)Nn * F_IN * 4, stream);
    k_edge_agg<HIDC><<<nblk((long long)ET, 8), 256, 0, stream>>>(srcv, dstv, E, Nn, HEADS,
                                                                 aS1, aD1, em1, dn1, h1, o1);
    k_fin_elu_bf<<<nblk((long long)Nn * F_IN, 256), 256, 0, stream>>>(o1, b1, h1b, Nn, F_IN);

    // ---------------- Layer 2: H=1, C=32, mean(=identity) ----------------
    k_transpose_cvt<<<nblk(F_IN * HIDC, 256), 256, 0, stream>>>(W2, W2t, F_IN, HIDC, HIDC);
    launch_gemm<F_IN, 32, 32, 32>(h1b, W2t, z2, Nn, stream);
    k_alpha<<<nblk((long long)Nn, 256), 256, 0, stream>>>(z2, as2, ad2, aS2, aD2, Nn, 1, HIDC);
    k_init_u32<<<nblk((long long)Nn, 256), 256, 0, stream>>>(em2, NEGINF_ORD, Nn);
    k_edge_max<<<nblk((long long)ET, 256), 256, 0, stream>>>(srcv, dstv, E, Nn, 1, aS2, aD2, em2);
    hipMemsetAsync(dn2, 0, (size_t)Nn * 4, stream);
    k_edge_sum<<<nblk((long long)ET, 256), 256, 0, stream>>>(srcv, dstv, E, Nn, 1, aS2, aD2, em2, dn2);
    hipMemsetAsync(o2, 0, (size_t)Nn * HIDC * 4, stream);
    k_edge_agg<HIDC><<<nblk((long long)ET, 8), 256, 0, stream>>>(srcv, dstv, E, Nn, 1,
                                                                 aS2, aD2, em2, dn2, z2, o2);
    k_fin_elu_bf<<<nblk((long long)Nn * HIDC, 256), 256, 0, stream>>>(o2, b2, z2b, Nn, HIDC);

    // ---------------- Layer 3: H=1, C=10 ----------------
    k_transpose_cvt<<<nblk(HIDC * 16, 256), 256, 0, stream>>>(W3, W3t, HIDC, OUTC, 16);
    launch_gemm<HIDC, 16, OUTC, OUTC>(z2b, W3t, h3, Nn, stream);
    k_alpha<<<nblk((long long)Nn, 256), 256, 0, stream>>>(h3, as3, ad3, aS3, aD3, Nn, 1, OUTC);
    k_init_u32<<<nblk((long long)Nn, 256), 256, 0, stream>>>(em3, NEGINF_ORD, Nn);
    k_edge_max<<<nblk((long long)ET, 256), 256, 0, stream>>>(srcv, dstv, E, Nn, 1, aS3, aD3, em3);
    hipMemsetAsync(dn3, 0, (size_t)Nn * 4, stream);
    k_edge_sum<<<nblk((long long)ET, 256), 256, 0, stream>>>(srcv, dstv, E, Nn, 1, aS3, aD3, em3, dn3);
    hipMemsetAsync(outf, 0, (size_t)Nn * OUTC * 4, stream);
    k_edge_agg<OUTC><<<nblk((long long)ET, 8), 256, 0, stream>>>(srcv, dstv, E, Nn, 1,
                                                                 aS3, aD3, em3, dn3, h3, outf);
    k_fin_bias<<<nblk((long long)Nn * OUTC, 256), 256, 0, stream>>>(outf, b3, Nn, OUTC);
}
